// TemporalLinearAttention_49864570307033
// MI455X (gfx1250) — matrix operand order repaired and checked by reading, not compiled, not run on hardware
//
#include <hip/hip_runtime.h>

// CDNA5 / gfx1250 fused temporal linear attention.
// f32 inputs -> f16 WMMA operands (v_wmma_f32_16x16x32_f16), f32 accumulation.
// Register-tiled 1x2 (A-fragment reuse across two N tiles), b128 fragment loads.

typedef __attribute__((ext_vector_type(16))) _Float16 v16h;
typedef __attribute__((ext_vector_type(8)))  _Float16 v8h;
typedef __attribute__((ext_vector_type(8)))  float    v8f;

// ---- CDNA5 16x16x32 f16 WMMA fragment layouts (wave32) ----
// A (16x32, lane=M):  half h -> K = h + 8*(h>=8) + 8*(lane>=16)   (two 8-elem runs)
// B (32x16, lane=N):  half h -> K = h + 8*(h>=8) + 8*(lane>=16)   (two 8-elem runs)
// C/D (16x16 f32):    vgpr r -> M = r + 8*(lane>=16), N = lane&15
__device__ __forceinline__ int kA16(int h, int hi) { return h + ((h & 8) ? 8 : 0) + (hi ? 8 : 0); }
__device__ __forceinline__ int kB16(int h, int hi) { return h + ((h & 8) ? 8 : 0) + (hi ? 8 : 0); }

// A fragment from row-major f32 global; vectorized as 4x float4 (global_load_b128).
__device__ __forceinline__ v16h loadA_g(const float* __restrict__ base, int row0, int ld, int kk, int lane) {
    const int M = lane & 15, hi = lane >> 4;
    const float4* p4 = reinterpret_cast<const float4*>(base + (size_t)(row0 + M) * ld + kk) + (hi ? 2 : 0);
    const float4 r0 = p4[0], r1 = p4[1];   // K = s .. s+7      (s = 8*hi)
    const float4 r2 = p4[4], r3 = p4[5];   // K = s+16 .. s+23
    v16h a;
    a[0]  = (_Float16)r0.x; a[1]  = (_Float16)r0.y; a[2]  = (_Float16)r0.z; a[3]  = (_Float16)r0.w;
    a[4]  = (_Float16)r1.x; a[5]  = (_Float16)r1.y; a[6]  = (_Float16)r1.z; a[7]  = (_Float16)r1.w;
    a[8]  = (_Float16)r2.x; a[9]  = (_Float16)r2.y; a[10] = (_Float16)r2.z; a[11] = (_Float16)r2.w;
    a[12] = (_Float16)r3.x; a[13] = (_Float16)r3.y; a[14] = (_Float16)r3.z; a[15] = (_Float16)r3.w;
    return a;
}

// B fragment from row-major f32 global: elem (K,N) = base[(kk+K)*ld + col0 + N]
__device__ __forceinline__ v16h loadB_g(const float* __restrict__ base, int kk, int ld, int col0, int lane) {
    v16h b; const int N = lane & 15, hi = lane >> 4;
    const float* p = base + (size_t)kk * ld + col0 + N;
#pragma unroll
    for (int h = 0; h < 16; ++h) b[h] = (_Float16)p[(size_t)kB16(h, hi) * ld];
    return b;
}

// A fragment from f16 rows: two contiguous 16B runs -> ds_load_b128 pair.
__device__ __forceinline__ v16h loadA_h_row(const _Float16* base, int row0, int ld, int kk, int lane) {
    const int M = lane & 15, hi = lane >> 4;
    const _Float16* p = base + (size_t)(row0 + M) * ld + kk + (hi ? 8 : 0);
    const v8h lo = *(const v8h*)p;          // K = s .. s+7
    const v8h hh = *(const v8h*)(p + 16);   // K = s+16 .. s+23
    v16h a;
#pragma unroll
    for (int h = 0; h < 8; ++h) { a[h] = lo[h]; a[8 + h] = hh[h]; }
    return a;
}

// A fragment, transposed access: elem (M,K) = base[(kk+K)*ld + row0 + M]
__device__ __forceinline__ v16h loadA_h_T(const _Float16* base, int row0, int ld, int kk, int lane) {
    v16h a; const int M = lane & 15, hi = lane >> 4;
#pragma unroll
    for (int h = 0; h < 16; ++h) a[h] = base[(size_t)(kk + kA16(h, hi)) * ld + row0 + M];
    return a;
}

// B fragment, K along rows: elem (K,N) = base[(kk+K)*ld + col0 + N]
__device__ __forceinline__ v16h loadB_h_rowK(const _Float16* base, int kk, int ld, int col0, int lane) {
    v16h b; const int N = lane & 15, hi = lane >> 4;
#pragma unroll
    for (int h = 0; h < 16; ++h) b[h] = base[(size_t)(kk + kB16(h, hi)) * ld + col0 + N];
    return b;
}

// B fragment, K contiguous within a row of the source (reads v[e][n] as B(K=n,N=e)):
// per-lane two 16B runs (K = s..s+7, s+16..s+23, s = 8*hi) -> ds_load_b128 pair.
__device__ __forceinline__ v16h loadB_h_colK(const _Float16* base, int col0, int ld, int kk, int lane) {
    const int N = lane & 15, hi = lane >> 4;
    const _Float16* p = base + (size_t)(col0 + N) * ld + kk + (hi ? 8 : 0);
    const v8h lo = *(const v8h*)p;
    const v8h hh = *(const v8h*)(p + 16);
    v16h b;
#pragma unroll
    for (int h = 0; h < 8; ++h) { b[h] = lo[h]; b[8 + h] = hh[h]; }
    return b;
}

#define WMMA_F16(A, B, C) \
    __builtin_amdgcn_wmma_f32_16x16x32_f16(false, (A), false, (B), (short)0, (C), false, false)

// -------------------------------------------------------------------------
// Kernel 1: one workgroup per (b*a, head).  8 wave32 / 256 threads.
// LDS: q,k,v [128x256] f16 + ctx [128x128] f16 + temb[384] + time[256] f32.
// -------------------------------------------------------------------------
__global__ void __launch_bounds__(256, 1)
tla_qkv_attn_kernel(const float* __restrict__ x,       // [ba][256(f)][256(t)]
                    const float* __restrict__ timev,   // [ba][256]
                    const float* __restrict__ w_qkv,   // [1536][256]
                    const float* __restrict__ w_time,  // [1536][256]
                    const float* __restrict__ b_time,  // [1536]
                    _Float16* __restrict__ hid)        // [ba][4][128][256] f16
{
    extern __shared__ char smem_raw[];
    _Float16* qT  = (_Float16*)smem_raw;        // 128*256
    _Float16* kT  = qT + 128 * 256;
    _Float16* vT  = kT + 128 * 256;
    _Float16* ctx = vT + 128 * 256;             // 128*128
    float* tembS  = (float*)(ctx + 128 * 128);  // 384
    float* timeS  = tembS + 384;                // 256

    const int tid  = threadIdx.x;
    const int lane = tid & 31;
    const int wave = tid >> 5;
    const int ba   = blockIdx.x >> 2;
    const int head = blockIdx.x & 3;

    const float* xba = x + (size_t)ba * 256 * 256;

    // ---- stage mish(time) ----
    {
        float xx = timev[(size_t)ba * 256 + tid];
        float sp = log1pf(__expf(xx));
        timeS[tid] = xx * tanhf(sp);
    }
    __syncthreads();

    // ---- temb for this WG's 384 output rows (q/k/v x 128) ----
    for (int i = tid; i < 384; i += 256) {
        const int g = i >> 7, lr = i & 127;
        const int orow = g * 512 + head * 128 + lr;
        const float* wr = w_time + (size_t)orow * 256;
        float s = b_time[orow];
        for (int e = 0; e < 256; ++e) s += timeS[e] * wr[e];
        tembS[i] = s;
    }
    __syncthreads();

    // ---- Step 1: qkv = W_slice[384x256] @ x[256x256] + temb -> LDS f16 ----
    // 24 row tiles x 8 column PAIRS (A fragment reused across two N tiles).
    for (int u = wave; u < 192; u += 8) {
        const int tr = u % 24, tcp = u / 24;
        const int g = tr >> 3;
        const int lr0 = (tr & 7) << 4;
        const int wrow0 = g * 512 + head * 128 + lr0;
        const int c0 = tcp * 32;
        v8f acc0 = {}, acc1 = {};
        for (int kk = 0; kk < 256; kk += 32) {
            if (kk + 32 < 256) __builtin_prefetch(xba + (size_t)(kk + 32) * 256 + c0 + lane, 0, 1);
            const v16h a  = loadA_g(w_qkv, wrow0, 256, kk, lane);
            const v16h b0 = loadB_g(xba, kk, 256, c0, lane);
            const v16h b1 = loadB_g(xba, kk, 256, c0 + 16, lane);
            acc0 = WMMA_F16(a, b0, acc0);
            acc1 = WMMA_F16(a, b1, acc1);
        }
        _Float16* dst = (g == 0) ? qT : (g == 1) ? kT : vT;
        const int hi = lane >> 4, N = lane & 15;
#pragma unroll
        for (int r = 0; r < 8; ++r) {
            const int m = lr0 + r + hi * 8;
            const float tb = tembS[g * 128 + m];
            dst[m * 256 + c0 + N]      = (_Float16)(acc0[r] + tb);
            dst[m * 256 + c0 + 16 + N] = (_Float16)(acc1[r] + tb);
        }
    }
    __syncthreads();

    // ---- Step 2: softmax over t for each of k's 128 rows ----
    if (tid < 128) {
        _Float16* row = kT + tid * 256;
        float mx = -1e30f;
        for (int n = 0; n < 256; ++n) mx = fmaxf(mx, (float)row[n]);
        float s = 0.f;
        for (int n = 0; n < 256; ++n) { float e = __expf((float)row[n] - mx); s += e; row[n] = (_Float16)e; }
        const float inv = 1.f / s;
        for (int n = 0; n < 256; ++n) row[n] = (_Float16)((float)row[n] * inv);
    }
    __syncthreads();

    // ---- Step 3: ctx[d][e] = sum_n k[d][n] * v[e][n]  (128x128) ----
    // 8 d tiles x 4 e PAIRS.
    for (int u = wave; u < 32; u += 8) {
        const int d0 = (u & 7) << 4, e0 = (u >> 3) << 5;
        v8f acc0 = {}, acc1 = {};
        for (int kk = 0; kk < 256; kk += 32) {
            const v16h a  = loadA_h_row(kT, d0, 256, kk, lane);
            const v16h b0 = loadB_h_colK(vT, e0, 256, kk, lane);
            const v16h b1 = loadB_h_colK(vT, e0 + 16, 256, kk, lane);
            acc0 = WMMA_F16(a, b0, acc0);
            acc1 = WMMA_F16(a, b1, acc1);
        }
        const int hi = lane >> 4, N = lane & 15;
#pragma unroll
        for (int r = 0; r < 8; ++r) {
            const int m = d0 + r + hi * 8;
            ctx[m * 128 + e0 + N]      = (_Float16)acc0[r];
            ctx[m * 128 + e0 + 16 + N] = (_Float16)acc1[r];
        }
    }
    __syncthreads();

    // ---- Step 4: out[e][n] = sum_d ctx[d][e] * q[d][n] -> hid (global f16) ----
    // 8 e tiles x 8 n PAIRS.
    _Float16* outp = hid + (size_t)(ba * 4 + head) * 128 * 256;
    for (int u = wave; u < 64; u += 8) {
        const int e0 = (u & 7) << 4, n0 = (u >> 3) << 5;
        v8f acc0 = {}, acc1 = {};
        for (int kk = 0; kk < 128; kk += 32) {
            const v16h a  = loadA_h_T(ctx, e0, 128, kk, lane);      // A(M=e,K=d) = ctx[d][e]
            const v16h b0 = loadB_h_rowK(qT, kk, 256, n0, lane);    // B(K=d,N=n) = q[d][n]
            const v16h b1 = loadB_h_rowK(qT, kk, 256, n0 + 16, lane);
            acc0 = WMMA_F16(a, b0, acc0);
            acc1 = WMMA_F16(a, b1, acc1);
        }
        const int hi = lane >> 4, N = lane & 15;
#pragma unroll
        for (int r = 0; r < 8; ++r) {
            const int m = e0 + r + hi * 8;
            outp[m * 256 + n0 + N]      = (_Float16)acc0[r];
            outp[m * 256 + n0 + 16 + N] = (_Float16)acc1[r];
        }
    }
}

// -------------------------------------------------------------------------
// Kernel 2: final projection per (b*a): out[256x256] = w_out[256x512] @ hid + b_out
// 16 o tiles x 8 n PAIRS per workgroup.
// -------------------------------------------------------------------------
__global__ void __launch_bounds__(256, 1)
tla_proj_kernel(const _Float16* __restrict__ hid,  // [ba][512][256] f16
                const float* __restrict__ w_out,   // [256][512]
                const float* __restrict__ b_out,   // [256]
                float* __restrict__ out)           // [ba][256][256]
{
    const int tid = threadIdx.x, lane = tid & 31, wave = tid >> 5;
    const int ba = blockIdx.x;
    const _Float16* hba = hid + (size_t)ba * 512 * 256;
    float* oba = out + (size_t)ba * 256 * 256;

    for (int u = wave; u < 128; u += 8) {
        const int o0 = (u & 15) << 4, n0 = (u >> 4) << 5;
        v8f acc0 = {}, acc1 = {};
        for (int kk = 0; kk < 512; kk += 32) {
            const v16h a  = loadA_g(w_out, o0, 512, kk, lane);
            const v16h b0 = loadB_h_rowK(hba, kk, 256, n0, lane);
            const v16h b1 = loadB_h_rowK(hba, kk, 256, n0 + 16, lane);
            acc0 = WMMA_F16(a, b0, acc0);
            acc1 = WMMA_F16(a, b1, acc1);
        }
        const int hi = lane >> 4, N = lane & 15;
#pragma unroll
        for (int r = 0; r < 8; ++r) {
            const int o = o0 + r + hi * 8;
            const float bo = b_out[o];
            oba[o * 256 + n0 + N]      = acc0[r] + bo;
            oba[o * 256 + n0 + 16 + N] = acc1[r] + bo;
        }
    }
}

extern "C" void kernel_launch(void* const* d_in, const int* in_sizes, int n_in,
                              void* d_out, int out_size, void* d_ws, size_t ws_size,
                              hipStream_t stream) {
    (void)in_sizes; (void)n_in; (void)out_size; (void)ws_size;
    const float* x      = (const float*)d_in[0];  // [64,8,256,256]
    const float* timev  = (const float*)d_in[1];  // [64,8,256]
    const float* w_qkv  = (const float*)d_in[2];  // [1536,256]
    const float* w_time = (const float*)d_in[3];  // [1536,256]
    const float* b_time = (const float*)d_in[4];  // [1536]
    const float* w_out  = (const float*)d_in[5];  // [256,512]
    const float* b_out  = (const float*)d_in[6];  // [256]
    float* out = (float*)d_out;
    _Float16* hid = (_Float16*)d_ws;              // 2048*128*256 f16 = 128 MiB scratch

    const size_t lds_bytes = (size_t)(3 * 128 * 256 + 128 * 128) * sizeof(_Float16)
                           + (size_t)(384 + 256) * sizeof(float); // 231,936 B (< 320 KB WGP LDS)
    hipFuncSetAttribute((const void*)tla_qkv_attn_kernel,
                        hipFuncAttributeMaxDynamicSharedMemorySize, (int)lds_bytes);

    tla_qkv_attn_kernel<<<dim3(64 * 8 * 4), dim3(256), lds_bytes, stream>>>(
        x, timev, w_qkv, w_time, b_time, hid);
    tla_proj_kernel<<<dim3(64 * 8), dim3(256), 0, stream>>>(hid, w_out, b_out, out);
}
